// PointPatchEmbed_5385888989213
// MI455X (gfx1250) — matrix operand order repaired and checked by reading, not compiled, not run on hardware
//
#include <hip/hip_runtime.h>
#include <hip/hip_fp16.h>
#include <math.h>

// Problem constants (match reference)
#define B_ 32
#define N_ 8192
#define G_ 128
#define K_ 32
#define H_ 128
#define D_ 384
#define NT_ (D_ / 16)   // 24 N-tiles
#define KC_ (H_ / 32)   // 4 k-steps of 32

typedef __attribute__((ext_vector_type(16))) _Float16 v16h;
typedef __attribute__((ext_vector_type(8)))  float    v8f;

// ---------------------------------------------------------------------------
// Kernel 0: repack W2 (f32, H x D row-major) into f16 WMMA B-fragment layout.
// B-matrix 32x16 (KxN) per v_wmma_f32_16x16x32_f16:
//   lane = (lane>=16 half) + N,  element e (0..15) -> K = (e>=8?16:0) + (lane>=16?8:0) + (e&7).
// Stored so each lane's 16 halves are contiguous (one 32B load per fragment):
//   w2f[(((kc*NT + nt)*32 + lane)*16 + e)]
// ---------------------------------------------------------------------------
__global__ void pack_w2_kernel(const float* __restrict__ W2,
                               _Float16* __restrict__ w2f) {
    int idx = blockIdx.x * blockDim.x + threadIdx.x;
    if (idx >= KC_ * NT_ * 32 * 16) return;
    int e    = idx & 15;
    int lane = (idx >> 4) & 31;
    int t    = idx >> 9;
    int nt   = t % NT_;
    int kc   = t / NT_;
    int r = kc * 32 + ((e >= 8) ? 16 : 0) + ((lane >= 16) ? 8 : 0) + (e & 7);  // K (row of W2)
    int c = nt * 16 + (lane & 15);                  // N (col of W2)
    w2f[idx] = (_Float16)W2[r * D_ + c];
}

// ---------------------------------------------------------------------------
// wave32 argmax/argmin reductions (lowest-index tie-break, matches jnp.argmax)
// ---------------------------------------------------------------------------
__device__ inline void wave_argmax(float& v, int& i) {
    #pragma unroll
    for (int off = 16; off > 0; off >>= 1) {
        float ov = __shfl_down(v, off, 32);
        int   oi = __shfl_down(i, off, 32);
        if (ov > v || (ov == v && oi < i)) { v = ov; i = oi; }
    }
}
__device__ inline void wave_argmin(float& v, int& i) {
    #pragma unroll
    for (int off = 16; off > 0; off >>= 1) {
        float ov = __shfl_down(v, off, 32);
        int   oi = __shfl_down(i, off, 32);
        if (ov < v || (ov == v && oi < i)) { v = ov; i = oi; }
    }
}

__device__ inline int block_argmax(float v, int i, float* rv, int* ri,
                                   int* bcast, int nwaves) {
    const int lane = threadIdx.x & 31, wave = threadIdx.x >> 5;
    wave_argmax(v, i);
    if (lane == 0) { rv[wave] = v; ri[wave] = i; }
    __syncthreads();
    if (threadIdx.x == 0) {
        float bv = rv[0]; int bi = ri[0];
        for (int w = 1; w < nwaves; ++w)
            if (rv[w] > bv || (rv[w] == bv && ri[w] < bi)) { bv = rv[w]; bi = ri[w]; }
        *bcast = bi;
    }
    __syncthreads();
    return *bcast;
}

__device__ inline int block_argmin(float v, int i, float* rv, int* ri,
                                   int* bcast, int nwaves) {
    const int lane = threadIdx.x & 31, wave = threadIdx.x >> 5;
    wave_argmin(v, i);
    if (lane == 0) { rv[wave] = v; ri[wave] = i; }
    __syncthreads();
    if (threadIdx.x == 0) {
        float bv = rv[0]; int bi = ri[0];
        for (int w = 1; w < nwaves; ++w)
            if (rv[w] < bv || (rv[w] == bv && ri[w] < bi)) { bv = rv[w]; bi = ri[w]; }
        *bcast = bi;
    }
    __syncthreads();
    return *bcast;
}

// ---------------------------------------------------------------------------
// Kernel 1: farthest point sampling. One block of 1024 threads (32 waves) per
// batch. Running min-distances live in 8 registers/thread (i = tid + s*1024);
// LDS holds only the 32-entry cross-wave reduction scratch.
// ---------------------------------------------------------------------------
__global__ __launch_bounds__(1024) void fps_kernel(
    const float* __restrict__ xyz,
    float* __restrict__ ws_centers,     // (B*G, 3)
    float* __restrict__ out_centers) {  // (B*G, 3)
    __shared__ float pr[3][32];
    __shared__ float rv[32];
    __shared__ int   ri[32];
    __shared__ int   bfar;

    const int b    = blockIdx.x;
    const int tid  = threadIdx.x;
    const int lane = tid & 31, wave = tid >> 5;
    const float* pts = xyz + (size_t)b * N_ * 3;

    // --- mean over N ---
    float sx = 0.f, sy = 0.f, sz = 0.f;
    #pragma unroll
    for (int s = 0; s < N_ / 1024; ++s) {
        const int i = tid + s * 1024;
        sx += pts[i * 3 + 0]; sy += pts[i * 3 + 1]; sz += pts[i * 3 + 2];
    }
    #pragma unroll
    for (int off = 16; off > 0; off >>= 1) {
        sx += __shfl_down(sx, off, 32);
        sy += __shfl_down(sy, off, 32);
        sz += __shfl_down(sz, off, 32);
    }
    if (lane == 0) { pr[0][wave] = sx; pr[1][wave] = sy; pr[2][wave] = sz; }
    __syncthreads();
    if (wave == 0) {
        float x = pr[0][lane], y = pr[1][lane], z = pr[2][lane];
        #pragma unroll
        for (int off = 16; off > 0; off >>= 1) {
            x += __shfl_down(x, off, 32);
            y += __shfl_down(y, off, 32);
            z += __shfl_down(z, off, 32);
        }
        if (lane == 0) {
            pr[0][0] = x / (float)N_; pr[1][0] = y / (float)N_; pr[2][0] = z / (float)N_;
        }
    }
    __syncthreads();
    const float mx = pr[0][0], my = pr[1][0], mz = pr[2][0];

    // --- initial farthest = argmax dist-to-mean; init running dist (regs) ---
    float regd[N_ / 1024];
    float bv = -1.f; int bi = 0;
    #pragma unroll
    for (int s = 0; s < N_ / 1024; ++s) {
        const int i = tid + s * 1024;
        float dx = pts[i * 3 + 0] - mx;
        float dy = pts[i * 3 + 1] - my;
        float dz = pts[i * 3 + 2] - mz;
        float d  = dx * dx + dy * dy + dz * dz;
        regd[s] = 1e10f;
        if (d > bv) { bv = d; bi = i; }
    }
    int far = block_argmax(bv, bi, rv, ri, &bfar, 32);

    // --- 128 dependent rounds; emit current farthest, then min-update ---
    for (int g = 0; g < G_; ++g) {
        const float cx = pts[far * 3 + 0];
        const float cy = pts[far * 3 + 1];
        const float cz = pts[far * 3 + 2];
        if (tid == 0) {
            const int o = b * G_ + g;
            ws_centers[o * 3 + 0] = cx; out_centers[o * 3 + 0] = cx;
            ws_centers[o * 3 + 1] = cy; out_centers[o * 3 + 1] = cy;
            ws_centers[o * 3 + 2] = cz; out_centers[o * 3 + 2] = cz;
        }
        float v = -1.f; int ix = 0;
        #pragma unroll
        for (int s = 0; s < N_ / 1024; ++s) {
            const int i = tid + s * 1024;
            float dx = pts[i * 3 + 0] - cx;
            float dy = pts[i * 3 + 1] - cy;
            float dz = pts[i * 3 + 2] - cz;
            float d  = dx * dx + dy * dy + dz * dz;
            float nd = fminf(regd[s], d);
            regd[s] = nd;
            if (nd > v) { v = nd; ix = i; }
        }
        far = block_argmax(v, ix, rv, ri, &bfar, 32);
    }
}

// ---------------------------------------------------------------------------
// Kernel 2: brute-force kNN (K=32 smallest of 8192) + rel_xyz precompute.
// One 256-thread block per (b,g). Each thread owns 32 candidate distances in
// registers (i = tid + s*256); 32 argmin rounds mask in-register. Near-zero
// LDS so WGP residency is wave-limited, not LDS-limited.
// ---------------------------------------------------------------------------
__global__ __launch_bounds__(256) void knn_kernel(
    const float* __restrict__ xyz,
    const float* __restrict__ ws_centers,
    int*   __restrict__ out_gidx,       // (B*G, K) int32 bits
    float* __restrict__ ws_rel) {       // (B*G*K, 3)
    __shared__ float rv[8];
    __shared__ int   ri[8];
    __shared__ int   bbest;

    const int bg  = blockIdx.x;
    const int b   = bg / G_;
    const int tid = threadIdx.x;
    const float* pts = xyz + (size_t)b * N_ * 3;

    const float cx = ws_centers[bg * 3 + 0];
    const float cy = ws_centers[bg * 3 + 1];
    const float cz = ws_centers[bg * 3 + 2];

    float d[N_ / 256];
    #pragma unroll
    for (int s = 0; s < N_ / 256; ++s) {
        const int i = tid + s * 256;
        float dx = pts[i * 3 + 0] - cx;
        float dy = pts[i * 3 + 1] - cy;
        float dz = pts[i * 3 + 2] - cz;
        d[s] = dx * dx + dy * dy + dz * dz;
    }

    for (int k = 0; k < K_; ++k) {
        float v = 3.4e38f; int ix = 0x7fffffff;
        #pragma unroll
        for (int s = 0; s < N_ / 256; ++s) {
            const int i = tid + s * 256;
            if (d[s] < v) { v = d[s]; ix = i; }
        }
        const int bii = block_argmin(v, ix, rv, ri, &bbest, 8);
        if (tid == 0) {
            out_gidx[bg * K_ + k] = bii;
            float* rel = ws_rel + ((size_t)bg * K_ + k) * 3;
            rel[0] = pts[bii * 3 + 0] - cx;
            rel[1] = pts[bii * 3 + 1] - cy;
            rel[2] = pts[bii * 3 + 2] - cz;
        }
        // in-register masking (exactly one (thread, slot) matches)
        #pragma unroll
        for (int s = 0; s < N_ / 256; ++s)
            if (bii == tid + s * 256) d[s] = 3.4e38f;
    }
}

// ---------------------------------------------------------------------------
// Kernel 3: fused MiniPointNet per group via WMMA.
//   h  = gelu(rel @ W1 + b1)            (32 x 128, f16 in LDS, A-fragment order)
//   f  = h @ W2                          (v_wmma_f32_16x16x32_f16, f32 acc)
//   tok= max over 32 rows + b2           (in-register + shfl_xor(16) epilogue)
// One 256-thread (8-wave) block per (b,g); each wave owns 3 N-tiles.
// A-fragments are preloaded once (loop-invariant); per N-tile all four k-step
// B-fragments are preloaded so the 8 WMMAs issue behind a single load wait.
//
// A-fragment (16x32 f16): lane<16 -> M=lane, lane>=16 -> M=lane-16;
//   element e -> K = (e>=8?16:0) + (lane>=16?8:0) + (e&7).
// C/D (16x16 f32): lane<16 -> M=vgpr, N=lane;  lane>=16 -> M=8+vgpr, N=lane-16.
// ---------------------------------------------------------------------------
__global__ __launch_bounds__(256) void mlp_kernel(
    const float*    __restrict__ ws_rel,
    const _Float16* __restrict__ w2f,
    const float*    __restrict__ W1,
    const float*    __restrict__ b1,
    const float*    __restrict__ b2,
    float*          __restrict__ out_tok) {
    __shared__ __attribute__((aligned(32))) _Float16 hfrag[2 * KC_ * 32 * 16]; // 8 KB
    __shared__ float W1s[3 * H_];
    __shared__ float b1s[H_];

    const int bg   = blockIdx.x;
    const int tid  = threadIdx.x;
    const int lane = tid & 31, wave = tid >> 5;

    if (tid < H_) {
        W1s[tid]          = W1[tid];
        W1s[H_ + tid]     = W1[H_ + tid];
        W1s[2 * H_ + tid] = W1[2 * H_ + tid];
        b1s[tid]          = b1[tid];
    }
    __syncthreads();

    // ---- layer 1 + exact GELU: each thread computes 16 h values ----
    const int m  = tid >> 3;          // point row 0..31
    const int jb = (tid & 7) << 4;    // hidden col base
    const float* rel = ws_rel + ((size_t)bg * K_ + m) * 3;
    const float rx = rel[0], ry = rel[1], rz = rel[2];
    const int mt = m >> 4, lm = m & 15;
    #pragma unroll
    for (int j = 0; j < 16; ++j) {
        const int col = jb + j;
        float x  = rx * W1s[col] + ry * W1s[H_ + col] + rz * W1s[2 * H_ + col] + b1s[col];
        float gx = 0.5f * x * (1.0f + erff(x * 0.70710678118654752f));
        const int kc    = col >> 5;
        const int ko    = col & 31;
        const int lhi   = (ko >> 3) & 1;
        const int e     = (((ko >> 4) & 1) << 3) | (ko & 7);
        const int flane = lm | (lhi << 4);
        hfrag[((mt * KC_ + kc) * 32 + flane) * 16 + e] = (_Float16)gx;
    }
    __syncthreads();

    // ---- preload all 8 A-fragments (loop-invariant across N-tiles) ----
    v16h a00 = *(const v16h*)(hfrag + ((0 * KC_ + 0) * 32 + lane) * 16);
    v16h a01 = *(const v16h*)(hfrag + ((0 * KC_ + 1) * 32 + lane) * 16);
    v16h a02 = *(const v16h*)(hfrag + ((0 * KC_ + 2) * 32 + lane) * 16);
    v16h a03 = *(const v16h*)(hfrag + ((0 * KC_ + 3) * 32 + lane) * 16);
    v16h a10 = *(const v16h*)(hfrag + ((1 * KC_ + 0) * 32 + lane) * 16);
    v16h a11 = *(const v16h*)(hfrag + ((1 * KC_ + 1) * 32 + lane) * 16);
    v16h a12 = *(const v16h*)(hfrag + ((1 * KC_ + 2) * 32 + lane) * 16);
    v16h a13 = *(const v16h*)(hfrag + ((1 * KC_ + 3) * 32 + lane) * 16);

    // ---- layer 2 via WMMA + K-max epilogue ----
    for (int nt = wave; nt < NT_; nt += 8) {
        const _Float16* bp = w2f + ((size_t)nt * 32 + lane) * 16;
        // preload all 4 k-step B-fragments -> one clause, one wait, 8 WMMAs
        v16h bf0 = *(const v16h*)(bp + (size_t)0 * NT_ * 32 * 16);
        v16h bf1 = *(const v16h*)(bp + (size_t)1 * NT_ * 32 * 16);
        v16h bf2 = *(const v16h*)(bp + (size_t)2 * NT_ * 32 * 16);
        v16h bf3 = *(const v16h*)(bp + (size_t)3 * NT_ * 32 * 16);

        v8f acc0 = {};   // rows 0..15
        v8f acc1 = {};   // rows 16..31
        acc0 = __builtin_amdgcn_wmma_f32_16x16x32_f16(false, a00, false, bf0, (short)0, acc0, false, false);
        acc1 = __builtin_amdgcn_wmma_f32_16x16x32_f16(false, a10, false, bf0, (short)0, acc1, false, false);
        acc0 = __builtin_amdgcn_wmma_f32_16x16x32_f16(false, a01, false, bf1, (short)0, acc0, false, false);
        acc1 = __builtin_amdgcn_wmma_f32_16x16x32_f16(false, a11, false, bf1, (short)0, acc1, false, false);
        acc0 = __builtin_amdgcn_wmma_f32_16x16x32_f16(false, a02, false, bf2, (short)0, acc0, false, false);
        acc1 = __builtin_amdgcn_wmma_f32_16x16x32_f16(false, a12, false, bf2, (short)0, acc1, false, false);
        acc0 = __builtin_amdgcn_wmma_f32_16x16x32_f16(false, a03, false, bf3, (short)0, acc0, false, false);
        acc1 = __builtin_amdgcn_wmma_f32_16x16x32_f16(false, a13, false, bf3, (short)0, acc1, false, false);

        // per-lane max over its 16 rows (8 from each M-tile), then cross-half
        float mxv = acc0[0];
        #pragma unroll
        for (int i = 1; i < 8; ++i) mxv = fmaxf(mxv, acc0[i]);
        #pragma unroll
        for (int i = 0; i < 8; ++i) mxv = fmaxf(mxv, acc1[i]);
        mxv = fmaxf(mxv, __shfl_xor(mxv, 16, 32));
        if (lane < 16) {
            const int col = nt * 16 + lane;
            out_tok[(size_t)bg * D_ + col] = mxv + b2[col];
        }
    }
}

// ---------------------------------------------------------------------------
// Launch. d_out layout (return order): tokens (B,G,D) f32 | centers (B,G,3)
// f32 | group_idx (B,G,K) int32-bits.
// ---------------------------------------------------------------------------
extern "C" void kernel_launch(void* const* d_in, const int* in_sizes, int n_in,
                              void* d_out, int out_size, void* d_ws, size_t ws_size,
                              hipStream_t stream) {
    (void)in_sizes; (void)n_in; (void)out_size; (void)ws_size;
    const float* xyz = (const float*)d_in[0];
    const float* W1  = (const float*)d_in[1];
    const float* b1  = (const float*)d_in[2];
    const float* W2  = (const float*)d_in[3];
    const float* b2  = (const float*)d_in[4];

    float* out      = (float*)d_out;
    float* out_tok  = out;                                       // B*G*D
    float* out_cen  = out + (size_t)B_ * G_ * D_;                // B*G*3
    int*   out_gid  = (int*)(out + (size_t)B_ * G_ * D_ + (size_t)B_ * G_ * 3);

    char* ws = (char*)d_ws;
    float*    ws_centers = (float*)(ws);                          //  49152 B
    float*    ws_rel     = (float*)(ws + 49152);                  // 1572864 B
    _Float16* w2f        = (_Float16*)(ws + 49152 + 1572864);     //   98304 B

    pack_w2_kernel<<<(KC_ * NT_ * 32 * 16 + 255) / 256, 256, 0, stream>>>(W2, w2f);
    fps_kernel<<<B_, 1024, 0, stream>>>(xyz, ws_centers, out_cen);
    knn_kernel<<<B_ * G_, 256, 0, stream>>>(xyz, ws_centers, out_gid, ws_rel);
    mlp_kernel<<<B_ * G_, 256, 0, stream>>>(ws_rel, w2f, W1, b1, b2, out_tok);
}
